// GCNEmbedding_75265006895359
// MI455X (gfx1250) — matrix operand order repaired and checked by reading, not compiled, not run on hardware
//
#include <hip/hip_runtime.h>
#include <hip/hip_bf16.h>

#define DF 128          // feature dim
#define WLDS_STRIDE 136 // 128 + 8 bf16 pad -> 272B row stride, kills LDS bank conflicts

typedef __attribute__((ext_vector_type(16))) __bf16 v16bf;
typedef __attribute__((ext_vector_type(8)))  float  v8f;

// ---------------- utility kernels ----------------

__global__ void gcn_zero4(float4* __restrict__ p, long long n4) {
    long long i = (long long)blockIdx.x * blockDim.x + threadIdx.x;
    if (i < n4) p[i] = make_float4(0.f, 0.f, 0.f, 0.f);
}

__global__ void gcn_degree(const int* __restrict__ src, const int* __restrict__ dst,
                           float* __restrict__ deg_out, float* __restrict__ deg_in, int nE) {
    int e = blockIdx.x * blockDim.x + threadIdx.x;
    if (e >= nE) return;
    atomicAdd(&deg_out[src[e]], 1.0f);
    atomicAdd(&deg_in [dst[e]], 1.0f);
}

// in-place deg -> deg^{-1/2} (deg==0 -> 1.0 -> rsqrt(1)=1)
__global__ void gcn_norm(float* __restrict__ a, float* __restrict__ b, int n) {
    int i = blockIdx.x * blockDim.x + threadIdx.x;
    if (i >= n) return;
    float x = a[i]; a[i] = rsqrtf(x > 0.f ? x : 1.f);
    float y = b[i]; b[i] = rsqrtf(y > 0.f ? y : 1.f);
}

__global__ void gcn_counts(const int* __restrict__ gids, float* __restrict__ counts, int n) {
    int i = blockIdx.x * blockDim.x + threadIdx.x;
    if (i >= n) return;
    atomicAdd(&counts[gids[i]], 1.0f);
}

// xs = x * norm_src[row]  (float4-vectorized; n4 = N*32)
__global__ void gcn_scale(const float4* __restrict__ x, const float* __restrict__ nsrc,
                          float4* __restrict__ xs, long long n4) {
    long long i = (long long)blockIdx.x * blockDim.x + threadIdx.x;
    if (i >= n4) return;
    float s = nsrc[i >> 5];
    float4 v = x[i];
    xs[i] = make_float4(v.x * s, v.y * s, v.z * s, v.w * s);
}

// one wave per edge: lanes 0..31 cover the 32 float4s of a 128-wide row.
// coalesced gather of xs[src], coalesced global_atomic_add_f32 into agg[dst].
// working set (xs + agg ~= 102 MB) fits the 192 MB L2, so default RT caching.
__global__ void gcn_scatter(const float4* __restrict__ xs4, const int* __restrict__ src,
                            const int* __restrict__ dst, float* __restrict__ agg, int nE) {
    long long gid = (long long)blockIdx.x * blockDim.x + threadIdx.x;
    int e = (int)(gid >> 5);
    int q = (int)(gid & 31);
    if (e >= nE) return;
    int s = src[e];
    int d = dst[e];
    float4 v = xs4[(size_t)s * 32 + q];
    float* p = agg + (size_t)d * DF + q * 4;
    atomicAdd(p + 0, v.x);
    atomicAdd(p + 1, v.y);
    atomicAdd(p + 2, v.z);
    atomicAdd(p + 3, v.w);
}

// ---------------- WMMA GEMM:  out = relu?((A*norm_dst) @ W + b) ----------------
// A: [n_rows][128] f32 (unscaled agg), W: [128][128] f32 row-major (k-major),
// bias: [128]. mode 0: store relu(h) to out[n_rows][128].
// mode 1: no relu; atomically pool rows into out[G][128] keyed by gids[row].
// Block = 256 threads = 8 waves; wave w handles row tile (blockIdx*8 + w)*16.
__global__ void __launch_bounds__(256)
gcn_gemm(const float* __restrict__ A, const float* __restrict__ ndst,
         const float* __restrict__ W, const float* __restrict__ bias,
         const int* __restrict__ gids, float* __restrict__ out,
         int n_rows, int mode) {
    // W staged transposed in LDS: WlT[col][k], bf16, padded stride
    __shared__ __bf16 WlT[DF * WLDS_STRIDE];

    int t = threadIdx.x;
    for (int i = t; i < DF * DF; i += 256) {
        int k = i >> 7;        // row of W
        int n = i & 127;       // col of W
        WlT[n * WLDS_STRIDE + k] = (__bf16)W[i];   // coalesced global f32 read
    }
    __syncthreads();

    int wave = t >> 5;
    int lane = t & 31;
    int half = lane >> 4;      // 0: lanes 0-15, 1: lanes 16-31
    int lcol = lane & 15;
    int m0 = (blockIdx.x * 8 + wave) * 16;
    if (m0 >= n_rows) return;  // wave-uniform exit

    // ---- A fragments (16x32 bf16 per kstep), row = m0 + lcol, scaled by norm_dst ----
    // layout: lane half h holds K = {ks*32 + 8h .. +7} and {ks*32 + 16 + 8h .. +7}
    int r = m0 + lcol;
    bool rv = (r < n_rows);
    float ns = rv ? ndst[r] : 0.f;
    const float* arow = A + (size_t)(rv ? r : 0) * DF;

    v16bf afrag[4];
#pragma unroll
    for (int ks = 0; ks < 4; ++ks) {
        int kb = ks * 32 + half * 8;
#pragma unroll
        for (int j = 0; j < 8; ++j) {
            afrag[ks][j]     = (__bf16)(arow[kb + j]      * ns);
            afrag[ks][j + 8] = (__bf16)(arow[kb + 16 + j] * ns);
        }
    }

    // ---- hoist per-row epilogue state out of the column-tile loop ----
    // C/D layout: VGPR v, lane -> (M = v + 8*half, N = lcol)
    bool rowok[8];
    size_t rbase[8];   // mode0: row*DF   mode1: gids[row]*DF
#pragma unroll
    for (int v = 0; v < 8; ++v) {
        int row = m0 + v + half * 8;
        rowok[v] = (row < n_rows);
        int rc = rowok[v] ? row : 0;
        rbase[v] = (mode == 0) ? (size_t)rc * DF : (size_t)gids[rc] * DF;
    }

    // ---- 8 output column tiles of 16 ----
#pragma unroll
    for (int nt = 0; nt < 8; ++nt) {
        int colg = nt * 16 + lcol;
        float bv = bias[colg];
        v8f acc;
#pragma unroll
        for (int v = 0; v < 8; ++v) acc[v] = bv;

#pragma unroll
        for (int ks = 0; ks < 4; ++ks) {
            // B fragment 32x16: lane (col=colg, half h) holds K = ks*32 + 8h .. +7 and ks*32 + 16 + 8h .. +7
            const __bf16* bp = &WlT[colg * WLDS_STRIDE + ks * 32 + half * 8];
            v16bf bfrag;
#pragma unroll
            for (int j = 0; j < 16; ++j) bfrag[j] = bp[j + (j & 8)];   // two 16B LDS reads

            acc = __builtin_amdgcn_wmma_f32_16x16x32_bf16(
                false, afrag[ks], false, bfrag, (short)0, acc, false, false);
        }

        if (mode == 0) {
#pragma unroll
            for (int v = 0; v < 8; ++v) {
                if (rowok[v]) out[rbase[v] + colg] = fmaxf(acc[v], 0.f);
            }
        } else {
#pragma unroll
            for (int v = 0; v < 8; ++v) {
                if (rowok[v]) atomicAdd(&out[rbase[v] + colg], acc[v]);
            }
        }
    }
}

// out[g][n] /= max(counts[g],1)
__global__ void gcn_finalize(float* __restrict__ out, const float* __restrict__ counts, int n) {
    int i = blockIdx.x * blockDim.x + threadIdx.x;
    if (i >= n) return;
    out[i] = out[i] / fmaxf(counts[i >> 7], 1.f);
}

// ---------------- host launcher ----------------

extern "C" void kernel_launch(void* const* d_in, const int* in_sizes, int n_in,
                              void* d_out, int out_size, void* d_ws, size_t ws_size,
                              hipStream_t stream) {
    const float* in_feat = (const float*)d_in[0];
    const float* W1      = (const float*)d_in[1];
    const float* b1      = (const float*)d_in[2];
    const float* W2      = (const float*)d_in[3];
    const float* b2      = (const float*)d_in[4];
    const int*   src     = (const int*)  d_in[5];
    const int*   dst     = (const int*)  d_in[6];
    const int*   gids    = (const int*)  d_in[7];
    float*       out     = (float*)d_out;

    const int N = in_sizes[0] / DF;
    const int E = in_sizes[5];
    // out_size = G * DF

    // workspace layout (floats)
    size_t Npad = ((size_t)N + 127) & ~(size_t)127;
    float* ws       = (float*)d_ws;
    float* norm_src = ws;                       // Npad
    float* norm_dst = ws + Npad;                // Npad
    float* counts   = ws + 2 * Npad;            // 128 slot (G<=128)
    float* xs       = counts + 128;             // Npad*128
    float* agg      = xs  + Npad * DF;          // Npad*128
    float* h        = agg + Npad * DF;          // Npad*128
    (void)ws_size; (void)n_in;

    const int B = 256;
    long long nfeat4 = (long long)N * (DF / 4);          // N*32 float4
    int zb_norm  = (int)((2 * Npad / 4 + B - 1) / B);
    int zb_feat  = (int)((nfeat4 + B - 1) / B);
    int eb       = (E + B - 1) / B;
    int nb       = (N + B - 1) / B;
    long long scat_thr = (long long)E * 32;
    int sb       = (int)((scat_thr + B - 1) / B);
    int gb       = (N + 127) / 128;                      // 8 waves/block, 16 rows/wave
    int ob4      = (out_size / 4 + B - 1) / B;
    int fb       = (out_size + B - 1) / B;

    // zero accumulation targets
    gcn_zero4<<<zb_norm, B, 0, stream>>>((float4*)norm_src, (long long)(2 * Npad / 4)); // degs
    gcn_zero4<<<1, B, 0, stream>>>((float4*)counts, 32);                                // counts
    gcn_zero4<<<ob4, B, 0, stream>>>((float4*)out, out_size / 4);                       // pooled out

    // degrees -> norms, graph counts
    gcn_degree<<<eb, B, 0, stream>>>(src, dst, norm_src, norm_dst, E);
    gcn_norm  <<<nb, B, 0, stream>>>(norm_src, norm_dst, N);
    gcn_counts<<<nb, B, 0, stream>>>(gids, counts, N);

    // ---- layer 1 ----
    gcn_scale  <<<zb_feat, B, 0, stream>>>((const float4*)in_feat, norm_src, (float4*)xs, nfeat4);
    gcn_zero4  <<<zb_feat, B, 0, stream>>>((float4*)agg, nfeat4);
    gcn_scatter<<<sb, B, 0, stream>>>((const float4*)xs, src, dst, agg, E);
    gcn_gemm   <<<gb, B, 0, stream>>>(agg, norm_dst, W1, b1, gids, h, N, /*mode=*/0);

    // ---- layer 2 (GEMM fused with graph mean-pool accumulation) ----
    gcn_scale  <<<zb_feat, B, 0, stream>>>((const float4*)h, norm_src, (float4*)xs, nfeat4);
    gcn_zero4  <<<zb_feat, B, 0, stream>>>((float4*)agg, nfeat4);
    gcn_scatter<<<sb, B, 0, stream>>>((const float4*)xs, src, dst, agg, E);
    gcn_gemm   <<<gb, B, 0, stream>>>(agg, norm_dst, W2, b2, gids, out, N, /*mode=*/1);

    // divide pooled sums by counts
    gcn_finalize<<<fb, B, 0, stream>>>(out, counts, out_size);
}
